// TGN_GAT_73246372266149
// MI455X (gfx1250) — matrix operand order repaired and checked by reading, not compiled, not run on hardware
//
#include <hip/hip_runtime.h>
#include <hip/hip_bf16.h>

#define HEADS 4
#define CH    64
#define OUTD  256   // HEADS*CH

typedef __attribute__((ext_vector_type(16))) _Float16 v16h;
typedef __attribute__((ext_vector_type(8)))  _Float16 v8h;
typedef __attribute__((ext_vector_type(8)))  float    v8f;

// ------------------------------------------------------------------
// f32 -> f16 packing (elementwise)
// ------------------------------------------------------------------
__global__ void pack_f16_kernel(const float* __restrict__ in, _Float16* __restrict__ out, int n) {
  int i = blockIdx.x * blockDim.x + threadIdx.x;
  if (i < n) out[i] = (_Float16)in[i];
}

// f32 W[K,Nout] -> f16 Wt[Nout,K] (transposed, so B fragments are contiguous)
__global__ void pack_wt_f16_kernel(const float* __restrict__ W, _Float16* __restrict__ Wt,
                                   int K, int Nout) {
  int i = blockIdx.x * blockDim.x + threadIdx.x;
  if (i >= K * Nout) return;
  int k = i / Nout, c = i - k * Nout;
  Wt[(size_t)c * K + k] = (_Float16)W[i];
}

// ------------------------------------------------------------------
// WMMA GEMM: C[M,NOUT] = A[M,K] @ W[K,NOUT], A f16 row-major, Wt f16 = W^T
// [NOUT,K] row-major. One wave = one 16x16 tile, fully unrolled K-loop of
// v_wmma_f32_16x16x32_f16, f32 accumulation. Fragment loads are b128:
//   A 16x32 f16 (ISA 7.12.2): lane L: m=L&15; k runs {kA0..kA0+7},{kA0+16..+23}
//   B 32x16 f16: lane L: n=L&15; k runs {kB0..kB0+7},{kB0+16..+23} in Wt row.
// Requires M%16==0 (50000 ok), K%32==0, NOUT%16==0.
// ------------------------------------------------------------------
template<int K, int NOUT>
__global__ void gemm_wmma_f16(const _Float16* __restrict__ A,
                              const _Float16* __restrict__ Wt,
                              float* __restrict__ C, int M) {
  const int wave = (int)((blockIdx.x * blockDim.x + threadIdx.x) >> 5);
  const int lane = (int)(threadIdx.x & 31);
  constexpr int tilesN = NOUT >> 4;
  const int nTiles = (M >> 4) * tilesN;
  if (wave >= nTiles) return;           // uniform per wave -> full EXEC inside
  const int tm = wave / tilesN;
  const int tn = wave - tm * tilesN;
  const int row0 = tm << 4;
  const int col0 = tn << 4;

  v8f acc = {};
  const int mA  = lane & 15;
  const int kA0 = (lane < 16) ? 0 : 8;
  const int nB  = lane & 15;
  const int kB0 = (lane < 16) ? 0 : 8;
  const _Float16* __restrict__ Arow = A  + (size_t)(row0 + mA) * K;
  const _Float16* __restrict__ Brow = Wt + (size_t)(col0 + nB) * K;

#pragma unroll
  for (int k0 = 0; k0 < K; k0 += 32) {
    v8h a0 = *(const v8h*)(Arow + k0 + kA0);        // global_load_b128
    v8h a1 = *(const v8h*)(Arow + k0 + kA0 + 16);
    v8h b0 = *(const v8h*)(Brow + k0 + kB0);
    v8h b1 = *(const v8h*)(Brow + k0 + kB0 + 16);
    v16h a = __builtin_shufflevector(a0, a1, 0,1,2,3,4,5,6,7,8,9,10,11,12,13,14,15);
    v16h b = __builtin_shufflevector(b0, b1, 0,1,2,3,4,5,6,7,8,9,10,11,12,13,14,15);
    acc = __builtin_amdgcn_wmma_f32_16x16x32_f16(false, a, false, b,
                                                 (short)0, acc, false, false);
  }
  // C/D f32 layout: vgpr r -> M = r (lanes<16) or 8+r (lanes>=16), N = lane&15
  const int mBase = (lane < 16) ? 0 : 8;
#pragma unroll
  for (int r = 0; r < 8; ++r)
    C[(size_t)(row0 + mBase + r) * NOUT + col0 + (lane & 15)] = acc[r];
}

// ------------------------------------------------------------------
// x[n,c] += proj_b[c] + type_emb[type[n], c]
// ------------------------------------------------------------------
__global__ void add_emb_kernel(float* __restrict__ x, const float* __restrict__ pb,
                               const float* __restrict__ emb, const int* __restrict__ types,
                               int n) {
  int i = blockIdx.x * blockDim.x + threadIdx.x;
  if (i >= n * CH) return;
  int node = i >> 6, c = i & 63;
  x[i] += pb[c] + emb[types[node] * CH + c];
}

// ------------------------------------------------------------------
// per (node,head) attention logits
// ------------------------------------------------------------------
__global__ void attn_kernel(const float* __restrict__ xp,
                            const float* __restrict__ att_s,
                            const float* __restrict__ att_d,
                            float* __restrict__ a_s, float* __restrict__ a_d, int n) {
  int idx = blockIdx.x * blockDim.x + threadIdx.x;   // over n*HEADS
  if (idx >= n * HEADS) return;
  int node = idx >> 2, h = idx & 3;
  const float* row = xp + (size_t)node * OUTD + h * CH;
  const float* as  = att_s + h * CH;
  const float* ad  = att_d + h * CH;
  float ss = 0.f, sd = 0.f;
#pragma unroll
  for (int c = 0; c < CH; ++c) { float v = row[c]; ss += v * as[c]; sd += v * ad[c]; }
  a_s[idx] = ss; a_d[idx] = sd;
}

// ------------------------------------------------------------------
__global__ void fill_f32(float* p, float v, int n) {
  int i = blockIdx.x * blockDim.x + threadIdx.x; if (i < n) p[i] = v;
}
__global__ void fill_i32(int* p, int v, int n) {
  int i = blockIdx.x * blockDim.x + threadIdx.x; if (i < n) p[i] = v;
}

// monotonic float<->int key so int atomicMax orders floats correctly
__device__ __forceinline__ int   fkey(float x) { int i = __float_as_int(x); return i < 0 ? (i ^ 0x7FFFFFFF) : i; }
__device__ __forceinline__ float funkey(int i) { return __int_as_float(i < 0 ? (i ^ 0x7FFFFFFF) : i); }
__device__ __forceinline__ float lrelu(float x) { return x > 0.f ? x : 0.2f * x; }

// ------------------------------------------------------------------
// segment max of leaky_relu(a_s[src]+a_d[dst]) over dst (incl. self loops)
// ------------------------------------------------------------------
__global__ void edge_max_kernel(const int* __restrict__ ei, int E, int n,
                                const float4* __restrict__ a_s4, const float4* __restrict__ a_d4,
                                int* __restrict__ segmax) {
  int e = blockIdx.x * blockDim.x + threadIdx.x;
  int Etot = E + n;
  if (e >= Etot) return;
  int src = (e < E) ? ei[e]     : (e - E);
  int dst = (e < E) ? ei[E + e] : (e - E);
  float4 s4 = a_s4[src];                 // b128
  float4 d4 = a_d4[dst];
  const float* sp = (const float*)&s4;
  const float* dp = (const float*)&d4;
#pragma unroll
  for (int h = 0; h < HEADS; ++h) {
    float s = lrelu(sp[h] + dp[h]);
    atomicMax(&segmax[dst * HEADS + h], fkey(s));
  }
}

// ------------------------------------------------------------------
// segment sum of exp(e - max) over dst
// ------------------------------------------------------------------
__global__ void edge_expsum_kernel(const int* __restrict__ ei, int E, int n,
                                   const float4* __restrict__ a_s4, const float4* __restrict__ a_d4,
                                   const int4* __restrict__ smax4, float* __restrict__ segsum) {
  int e = blockIdx.x * blockDim.x + threadIdx.x;
  int Etot = E + n;
  if (e >= Etot) return;
  int src = (e < E) ? ei[e]     : (e - E);
  int dst = (e < E) ? ei[E + e] : (e - E);
  float4 s4 = a_s4[src];
  float4 d4 = a_d4[dst];
  int4   m4 = smax4[dst];
  const float* sp = (const float*)&s4;
  const float* dp = (const float*)&d4;
  const int*   mp = (const int*)&m4;
#pragma unroll
  for (int h = 0; h < HEADS; ++h) {
    float s = lrelu(sp[h] + dp[h]);
    atomicAdd(&segsum[dst * HEADS + h], __expf(s - funkey(mp[h])));
  }
}

// ------------------------------------------------------------------
// wave-per-edge weighted scatter: agg[dst,h,:] += alpha[h] * xp[src,h,:]
// 32 lanes x 2 float4 b128 loads cover the 256-channel payload; 8 L2
// float atomic adds per lane. Prefetch the gathered row while the
// per-head softmax weights are computed.
// ------------------------------------------------------------------
__global__ void edge_scatter_kernel(const int* __restrict__ ei, int E, int n,
                                    const float4* __restrict__ a_s4, const float4* __restrict__ a_d4,
                                    const int4* __restrict__ smax4, const float4* __restrict__ ssum4,
                                    const float* __restrict__ xp, float* __restrict__ agg) {
  int wave = (int)((blockIdx.x * blockDim.x + threadIdx.x) >> 5);
  int lane = (int)(threadIdx.x & 31);
  int Etot = E + n;
  if (wave >= Etot) return;
  int e = wave;
  int src = (e < E) ? ei[e]     : (e - E);
  int dst = (e < E) ? ei[E + e] : (e - E);

  const float* __restrict__ xr = xp  + (size_t)src * OUTD;
  float* __restrict__       ar = agg + (size_t)dst * OUTD;
  __builtin_prefetch(xr + lane * 8, 0, 0);           // global_prefetch_b8

  float4 s4 = a_s4[src];
  float4 d4 = a_d4[dst];
  int4   m4 = smax4[dst];
  float4 q4 = ssum4[dst];
  const float* sp = (const float*)&s4;
  const float* dp = (const float*)&d4;
  const int*   mp = (const int*)&m4;
  const float* qp = (const float*)&q4;
  float alpha[HEADS];
#pragma unroll
  for (int h = 0; h < HEADS; ++h) {
    float s = lrelu(sp[h] + dp[h]);
    alpha[h] = __expf(s - funkey(mp[h])) / (qp[h] + 1e-16f);
  }
#pragma unroll
  for (int t = 0; t < 2; ++t) {
    int c = lane * 4 + t * 128;                      // 16B-aligned, head-uniform chunk
    float4 v = *(const float4*)(xr + c);             // global_load_b128
    float al = alpha[c >> 6];
    atomicAdd(&ar[c + 0], v.x * al);                 // GLOBAL_ATOMIC_ADD_F32 -> L2
    atomicAdd(&ar[c + 1], v.y * al);
    atomicAdd(&ar[c + 2], v.z * al);
    atomicAdd(&ar[c + 3], v.w * al);
  }
}

// wave32 all-lanes sum
__device__ __forceinline__ float wave_sum(float v) {
#pragma unroll
  for (int off = 16; off > 0; off >>= 1) v += __shfl_xor(v, off, 32);
  return v;
}

// ------------------------------------------------------------------
// layer0 epilogue: h = relu(LN_256(agg + bias0))
// ------------------------------------------------------------------
__global__ void ln0_kernel(const float* __restrict__ agg, const float* __restrict__ bias,
                           const float* __restrict__ g, const float* __restrict__ b,
                           float* __restrict__ hout, int n) {
  int wave = (int)((blockIdx.x * blockDim.x + threadIdx.x) >> 5);
  int lane = (int)(threadIdx.x & 31);
  if (wave >= n) return;
  const float* row = agg + (size_t)wave * OUTD;
  float v[8]; float mean = 0.f;
#pragma unroll
  for (int t = 0; t < 8; ++t) { v[t] = row[lane + 32 * t] + bias[lane + 32 * t]; mean += v[t]; }
  mean = wave_sum(mean) * (1.f / 256.f);
  float var = 0.f;
#pragma unroll
  for (int t = 0; t < 8; ++t) { float d = v[t] - mean; var += d * d; }
  var = wave_sum(var) * (1.f / 256.f);
  float rs = rsqrtf(var + 1e-5f);
  float* out = hout + (size_t)wave * OUTD;
#pragma unroll
  for (int t = 0; t < 8; ++t) {
    int c = lane + 32 * t;
    float y = (v[t] - mean) * rs * g[c] + b[c];
    out[c] = fmaxf(y, 0.f);
  }
}

// ------------------------------------------------------------------
// layer1 epilogue: out = relu(LN_64(mean_heads(agg) + bias1))
// ------------------------------------------------------------------
__global__ void ln1_kernel(const float* __restrict__ agg, const float* __restrict__ bias,
                           const float* __restrict__ g, const float* __restrict__ b,
                           float* __restrict__ out, int n) {
  int wave = (int)((blockIdx.x * blockDim.x + threadIdx.x) >> 5);
  int lane = (int)(threadIdx.x & 31);
  if (wave >= n) return;
  const float* row = agg + (size_t)wave * OUTD;
  float v[2]; float mean = 0.f;
#pragma unroll
  for (int t = 0; t < 2; ++t) {
    int c = lane + 32 * t;
    v[t] = (row[c] + row[c + 64] + row[c + 128] + row[c + 192]) * 0.25f + bias[c];
    mean += v[t];
  }
  mean = wave_sum(mean) * (1.f / 64.f);
  float var = 0.f;
#pragma unroll
  for (int t = 0; t < 2; ++t) { float d = v[t] - mean; var += d * d; }
  var = wave_sum(var) * (1.f / 64.f);
  float rs = rsqrtf(var + 1e-5f);
  float* o = out + (size_t)wave * CH;
#pragma unroll
  for (int t = 0; t < 2; ++t) {
    int c = lane + 32 * t;
    float y = (v[t] - mean) * rs * g[c] + b[c];
    o[c] = fmaxf(y, 0.f);
  }
}

// ------------------------------------------------------------------
static inline int cdiv(long a, long b) { return (int)((a + b - 1) / b); }

extern "C" void kernel_launch(void* const* d_in, const int* in_sizes, int n_in,
                              void* d_out, int out_size, void* d_ws, size_t ws_size,
                              hipStream_t stream) {
  const float* node_features = (const float*)d_in[0];
  const int*   node_types    = (const int*)  d_in[1];
  const int*   edge_index    = (const int*)  d_in[2];
  const float* type_emb      = (const float*)d_in[3];
  const float* proj_W        = (const float*)d_in[4];
  const float* proj_b        = (const float*)d_in[5];
  const float* W0            = (const float*)d_in[6];
  const float* att_src0      = (const float*)d_in[7];
  const float* att_dst0      = (const float*)d_in[8];
  const float* bias0         = (const float*)d_in[9];
  const float* ln0_g         = (const float*)d_in[10];
  const float* ln0_b         = (const float*)d_in[11];
  const float* W1            = (const float*)d_in[12];
  const float* att_src1      = (const float*)d_in[13];
  const float* att_dst1      = (const float*)d_in[14];
  const float* bias1         = (const float*)d_in[15];
  const float* ln1_g         = (const float*)d_in[16];
  const float* ln1_b         = (const float*)d_in[17];

  const int n = in_sizes[1];          // N = 50000
  const int E = in_sizes[2] / 2;      // 800000
  const int Etot = E + n;

  // workspace layout
  char* w = (char*)d_ws;
  float* x    = (float*)w; w += (size_t)n * CH   * 4;     // [N,64]
  float* xp   = (float*)w; w += (size_t)n * OUTD * 4;     // [N,256]
  float* agg  = (float*)w; w += (size_t)n * OUTD * 4;     // [N,256]
  float* hbuf = (float*)w; w += (size_t)n * OUTD * 4;     // [N,256]
  float* a_s  = (float*)w; w += (size_t)n * HEADS * 4;
  float* a_d  = (float*)w; w += (size_t)n * HEADS * 4;
  int*   smax = (int*)  w; w += (size_t)n * HEADS * 4;
  float* ssum = (float*)w; w += (size_t)n * HEADS * 4;
  _Float16* ah = (_Float16*)w; w += (size_t)n * OUTD * 2; // f16 packed activations
  _Float16* wt = (_Float16*)w; w += (size_t)OUTD * OUTD * 2; // f16 W^T
  (void)ws_size; (void)n_in; (void)out_size;

  const int T = 256;                  // 8 waves/block
  const int INT_NEG = (int)0x80000000;

  const float4* a_s4  = (const float4*)a_s;
  const float4* a_d4  = (const float4*)a_d;
  const int4*   smax4 = (const int4*)smax;
  const float4* ssum4 = (const float4*)ssum;

  // ---- input projection: x = nf @ proj_W  (+ proj_b + type_emb)
  {
    pack_f16_kernel<<<cdiv((long)n * CH, T), T, 0, stream>>>(node_features, ah, n * CH);
    pack_wt_f16_kernel<<<cdiv(CH * CH, T), T, 0, stream>>>(proj_W, wt, CH, CH);
    int tiles = (n >> 4) * (CH >> 4);
    gemm_wmma_f16<CH, CH><<<cdiv(tiles, 8), T, 0, stream>>>(ah, wt, x, n);
    add_emb_kernel<<<cdiv((long)n * CH, T), T, 0, stream>>>(x, proj_b, type_emb, node_types, n);
  }

  // ================= layer 0 =================
  {
    pack_f16_kernel<<<cdiv((long)n * CH, T), T, 0, stream>>>(x, ah, n * CH);
    pack_wt_f16_kernel<<<cdiv(CH * OUTD, T), T, 0, stream>>>(W0, wt, CH, OUTD);
    int tiles = (n >> 4) * (OUTD >> 4);
    gemm_wmma_f16<CH, OUTD><<<cdiv(tiles, 8), T, 0, stream>>>(ah, wt, xp, n);
    attn_kernel<<<cdiv((long)n * HEADS, T), T, 0, stream>>>(xp, att_src0, att_dst0, a_s, a_d, n);

    fill_i32<<<cdiv((long)n * HEADS, T), T, 0, stream>>>(smax, INT_NEG, n * HEADS);
    fill_f32<<<cdiv((long)n * HEADS, T), T, 0, stream>>>(ssum, 0.f, n * HEADS);
    fill_f32<<<cdiv((long)n * OUTD, T), T, 0, stream>>>(agg, 0.f, n * OUTD);

    edge_max_kernel   <<<cdiv(Etot, T), T, 0, stream>>>(edge_index, E, n, a_s4, a_d4, smax);
    edge_expsum_kernel<<<cdiv(Etot, T), T, 0, stream>>>(edge_index, E, n, a_s4, a_d4, smax4, ssum);
    edge_scatter_kernel<<<cdiv(Etot, 8), T, 0, stream>>>(edge_index, E, n, a_s4, a_d4, smax4, ssum4, xp, agg);

    ln0_kernel<<<cdiv(n, 8), T, 0, stream>>>(agg, bias0, ln0_g, ln0_b, hbuf, n);
  }

  // ================= layer 1 =================
  {
    pack_f16_kernel<<<cdiv((long)n * OUTD, T), T, 0, stream>>>(hbuf, ah, n * OUTD);
    pack_wt_f16_kernel<<<cdiv(OUTD * OUTD, T), T, 0, stream>>>(W1, wt, OUTD, OUTD);
    int tiles = (n >> 4) * (OUTD >> 4);
    gemm_wmma_f16<OUTD, OUTD><<<cdiv(tiles, 8), T, 0, stream>>>(ah, wt, xp, n);
    attn_kernel<<<cdiv((long)n * HEADS, T), T, 0, stream>>>(xp, att_src1, att_dst1, a_s, a_d, n);

    fill_i32<<<cdiv((long)n * HEADS, T), T, 0, stream>>>(smax, INT_NEG, n * HEADS);
    fill_f32<<<cdiv((long)n * HEADS, T), T, 0, stream>>>(ssum, 0.f, n * HEADS);
    fill_f32<<<cdiv((long)n * OUTD, T), T, 0, stream>>>(agg, 0.f, n * OUTD);

    edge_max_kernel   <<<cdiv(Etot, T), T, 0, stream>>>(edge_index, E, n, a_s4, a_d4, smax);
    edge_expsum_kernel<<<cdiv(Etot, T), T, 0, stream>>>(edge_index, E, n, a_s4, a_d4, smax4, ssum);
    edge_scatter_kernel<<<cdiv(Etot, 8), T, 0, stream>>>(edge_index, E, n, a_s4, a_d4, smax4, ssum4, xp, agg);

    ln1_kernel<<<cdiv(n, 8), T, 0, stream>>>(agg, bias1, ln1_g, ln1_b, (float*)d_out, n);
  }
}
